// SASTGCN_27943057228053
// MI455X (gfx1250) — matrix operand order repaired and checked by reading, not compiled, not run on hardware
//
#include <hip/hip_runtime.h>
#include <hip/hip_bf16.h>

// ---------------------------------------------------------------------------
// SASTGCN forward on MI455X (gfx1250): all matmuls via v_wmma_f32_16x16x32_bf16
// fp32 operands are converted to bf16 (RNE) while staging tiles in LDS;
// accumulation stays fp32 in the WMMA accumulator.
//
// All GEMM shapes in this network satisfy M%64==0, N%128==0, K%32==0, so the
// kernel uses exact tiling with NO bounds checks: float4 global loads
// (global_load_b128), packed bf16 ds_store_b64, and b128 LDS fragment loads.
// Each wave32 computes a 32x32 output tile = 4 independent WMMAs per K-step.
// ---------------------------------------------------------------------------

typedef __attribute__((ext_vector_type(16))) __bf16       v16bf;
typedef __attribute__((ext_vector_type(8)))  float        v8f;
typedef __attribute__((ext_vector_type(4)))  float        v4f;
typedef __attribute__((ext_vector_type(4)))  unsigned int v4u;
typedef __attribute__((ext_vector_type(2)))  unsigned int v2u;

#define GEMM_RELU    1
#define GEMM_OPB_T   2   // B stored as [N,K] row-major (weight.T multiply)
#define GEMM_BIAS_N  4   // bias indexed by output column (else by output row)
#define GEMM_STORE_T 8   // store C transposed: C[n*ldc + m]
#define GEMM_ACCUM  16   // C += Cin (same layout as normal store)

__device__ __forceinline__ unsigned short f2bf(float f) {
  unsigned u = __builtin_bit_cast(unsigned, f);
  u += 0x7FFFu + ((u >> 16) & 1u);          // round-to-nearest-even
  return (unsigned short)(u >> 16);
}

__device__ __forceinline__ v2u pack4(v4f v) {
  v2u p;
  p[0] = (unsigned)f2bf(v[0]) | ((unsigned)f2bf(v[1]) << 16);
  p[1] = (unsigned)f2bf(v[2]) | ((unsigned)f2bf(v[3]) << 16);
  return p;
}

union FragU { v4u q[2]; v16bf v; };

// C[M,N] = act( A[M,K] * B + bias ); B is [K,N] ('N') or [N,K] ('T').
// Block = 256 threads = 8 wave32s arranged 2(M) x 4(N); wave tile 32x32;
// block tile 64(M) x 128(N); K stepped by 32. Batched via blockIdx.z.
__global__ __launch_bounds__(256) void k_gemm_wmma(
    const float* __restrict__ A, int lda, long strideA,
    const float* __restrict__ Bm, int ldb, long strideB,
    const float* __restrict__ bias,
    const float* __restrict__ Cin,
    float* __restrict__ Cout, int ldc, long strideC,
    int K, int flags)
{
  __shared__ unsigned short sA[64][40];     // [m][k] bf16, padded stride
  __shared__ unsigned short sBt[128][40];   // [n][k] bf16, padded stride

  const int z = blockIdx.z;
  A    += (long)z * strideA;
  Bm   += (long)z * strideB;
  Cout += (long)z * strideC;
  const float* CinB = Cin ? (Cin + (long)z * strideC) : nullptr;

  const int tid  = threadIdx.x;
  const int wave = tid >> 5, lane = tid & 31;
  const int hi   = lane >> 4, ls = lane & 15;
  const int wm   = (wave >> 2) << 5;        // 0 / 32
  const int wn   = (wave & 3) << 5;         // 0 / 32 / 64 / 96
  const int m_blk = blockIdx.y * 64;
  const int n_blk = blockIdx.x * 128;
  const bool opbT = (flags & GEMM_OPB_T) != 0;

  v8f acc[2][2] = {};

  for (int kk = 0; kk < K; kk += 32) {
    // ---- A tile 64x32: 512 float4 loads, 2 per thread, packed b64 stores ---
    if (kk + 32 < K)   // uniform-branch prefetch of next K-slab
      __builtin_prefetch(A + (long)(m_blk + (tid >> 2)) * lda + kk + 32, 0, 1);
    #pragma unroll
    for (int u = 0; u < 2; ++u) {
      int idx4 = tid * 2 + u;
      int row  = idx4 >> 3;                 // 0..63
      int kq   = (idx4 & 7) << 2;           // 0..28
      v4f v = *(const v4f*)(A + (long)(m_blk + row) * lda + kk + kq);
      *(v2u*)&sA[row][kq] = pack4(v);
    }
    // ---- B tile 32(K)x128(N) into sBt[n][k]: 1024 float4, 4 per thread ----
    if (opbT) {                             // B[N,K]: contiguous in k
      #pragma unroll
      for (int u = 0; u < 4; ++u) {
        int idx4 = tid * 4 + u;
        int n  = idx4 >> 3;                 // 0..127
        int kq = (idx4 & 7) << 2;           // 0..28
        v4f v = *(const v4f*)(Bm + (long)(n_blk + n) * ldb + kk + kq);
        *(v2u*)&sBt[n][kq] = pack4(v);
      }
    } else {                                // B[K,N]: transpose-scatter
      #pragma unroll
      for (int u = 0; u < 4; ++u) {
        int idx4 = tid * 4 + u;
        int kq = idx4 >> 5;                 // 0..31
        int nq = (idx4 & 31) << 2;          // 0..124
        v4f v = *(const v4f*)(Bm + (long)(kk + kq) * ldb + n_blk + nq);
        #pragma unroll
        for (int i = 0; i < 4; ++i) sBt[nq + i][kq] = f2bf(v[i]);
      }
    }
    __syncthreads();

    // ---- fragments: two contiguous 16B runs each => ds_load_b128 x2 ------
    // A lane layout (16x32 bf16): e<8 -> k = hi*8+e ; e>=8 -> k = 16+hi*8+e-8
    // B lane layout (32x16 bf16): e<8 -> k = hi*8+e ; e>=8 -> k = 16+hi*8+e-8
    FragU fa0, fa1, fb0, fb1;
    const int ra0 = wm + ls, ra1 = wm + 16 + ls;
    const int nb0 = wn + ls, nb1 = wn + 16 + ls;
    fa0.q[0] = *(const v4u*)&sA[ra0][hi << 3];
    fa0.q[1] = *(const v4u*)&sA[ra0][16 + (hi << 3)];
    fa1.q[0] = *(const v4u*)&sA[ra1][hi << 3];
    fa1.q[1] = *(const v4u*)&sA[ra1][16 + (hi << 3)];
    fb0.q[0] = *(const v4u*)&sBt[nb0][hi << 3];
    fb0.q[1] = *(const v4u*)&sBt[nb0][16 + (hi << 3)];
    fb1.q[0] = *(const v4u*)&sBt[nb1][hi << 3];
    fb1.q[1] = *(const v4u*)&sBt[nb1][16 + (hi << 3)];

    // 4 independent WMMAs: no RAW chain between them (distinct accumulators)
    acc[0][0] = __builtin_amdgcn_wmma_f32_16x16x32_bf16(
        false, fa0.v, false, fb0.v, (short)0, acc[0][0], false, false);
    acc[0][1] = __builtin_amdgcn_wmma_f32_16x16x32_bf16(
        false, fa0.v, false, fb1.v, (short)0, acc[0][1], false, false);
    acc[1][0] = __builtin_amdgcn_wmma_f32_16x16x32_bf16(
        false, fa1.v, false, fb0.v, (short)0, acc[1][0], false, false);
    acc[1][1] = __builtin_amdgcn_wmma_f32_16x16x32_bf16(
        false, fa1.v, false, fb1.v, (short)0, acc[1][1], false, false);
    __syncthreads();
  }

  // ---- epilogue: C/D layout lane L: N = L&15, M = r + 8*(L>=16) ----------
  #pragma unroll
  for (int i = 0; i < 2; ++i) {
    #pragma unroll
    for (int j = 0; j < 2; ++j) {
      const int bm = m_blk + wm + (i << 4) + (hi << 3);
      const int gn = n_blk + wn + (j << 4) + ls;
      #pragma unroll
      for (int r = 0; r < 8; ++r) {
        int gm = bm + r;
        float v = acc[i][j][r];
        if (flags & GEMM_ACCUM) v += CinB[(long)gm * ldc + gn];
        if (bias) v += (flags & GEMM_BIAS_N) ? bias[gn] : bias[gm];
        if (flags & GEMM_RELU) v = v > 0.0f ? v : 0.0f;
        if (flags & GEMM_STORE_T) Cout[(long)gn * ldc + gm] = v;
        else                      Cout[(long)gm * ldc + gn] = v;
      }
    }
  }
}

// ---------------------------------------------------------------------------
// Elementwise helpers
// ---------------------------------------------------------------------------
__device__ __forceinline__ float sigmoidf_(float x) {
  return 1.0f / (1.0f + __expf(-x));
}

// gates: [64,2048] (i,f,g,o chunks of 512). c,h: [64,512] updated in place.
// If hseq != nullptr, also writes h into hseq[b, t_rel, :] ([64,12,512]).
__global__ void k_lstm_cell(const float* __restrict__ gates,
                            float* __restrict__ c, float* __restrict__ h,
                            float* __restrict__ hseq, int t_rel)
{
  int i = blockIdx.x * blockDim.x + threadIdx.x;
  if (i >= 64 * 512) return;
  int b = i >> 9, j = i & 511;
  const float* g = gates + b * 2048;
  float ig = sigmoidf_(g[j]);
  float fg = sigmoidf_(g[512 + j]);
  float gg = tanhf(g[1024 + j]);
  float og = sigmoidf_(g[1536 + j]);
  float cn = fg * c[i] + ig * gg;
  c[i] = cn;
  float hn = og * tanhf(cn);
  h[i] = hn;
  if (hseq) hseq[b * (12 * 512) + t_rel * 512 + j] = hn;
}

// feats[s,n,b] : b<64 -> x[b,s,n]; b>=64 -> x[b-64,s,n] - auto[b-64,s,n]
__global__ void k_build_feats(const float* __restrict__ x,
                              const float* __restrict__ auto_out,
                              float* __restrict__ feats)
{
  int i = blockIdx.x * blockDim.x + threadIdx.x;
  const int SN = 512 * 128;
  if (i >= 24 * SN) return;
  int s = i / SN;
  int r = i - s * SN;
  int n = r >> 7;
  int b = r & 127;
  float v;
  if (b < 64) {
    v = x[b * 12288 + s * 512 + n];
  } else {
    int off = (b - 64) * 12288 + s * 512 + n;
    v = x[off] - auto_out[off];
  }
  feats[i] = v;
}

// nf[64,2560] = concat(st_out[64,1536], hidden[64,1024])
__global__ void k_concat_nf(const float* __restrict__ st,
                            const float* __restrict__ hid,
                            float* __restrict__ nf)
{
  int i = blockIdx.x * blockDim.x + threadIdx.x;
  if (i >= 64 * 2560) return;
  int b = i / 2560, cc = i - b * 2560;
  nf[i] = (cc < 1536) ? st[b * 1536 + cc] : hid[b * 1024 + (cc - 1536)];
}

// ---------------------------------------------------------------------------
// Host orchestration
// ---------------------------------------------------------------------------
static inline void gemm(hipStream_t s,
                        const float* A, int lda, long sa,
                        const float* B, int ldb, long sb,
                        const float* bias, const float* Cin,
                        float* C, int ldc, long sc,
                        int M, int N, int K, int flags, int batch)
{
  dim3 grid((unsigned)(N / 128), (unsigned)(M / 64), (unsigned)batch);
  k_gemm_wmma<<<grid, dim3(256), 0, s>>>(A, lda, sa, B, ldb, sb, bias, Cin,
                                         C, ldc, sc, K, flags);
}

extern "C" void kernel_launch(void* const* d_in, const int* in_sizes, int n_in,
                              void* d_out, int out_size, void* d_ws, size_t ws_size,
                              hipStream_t stream) {
  (void)in_sizes; (void)n_in; (void)out_size; (void)ws_size;

  const float* x      = (const float*)d_in[0];   // [64,24,512]
  const float* adj    = (const float*)d_in[1];   // [512,512]
  const float* h0     = (const float*)d_in[2];   // [2,64,512]
  const float* c0     = (const float*)d_in[3];   // [2,64,512]
  const float* fc11_W = (const float*)d_in[4];   const float* fc11_b = (const float*)d_in[5];
  const float* fc12_W = (const float*)d_in[6];   const float* fc12_b = (const float*)d_in[7];
  const float* fc21_W = (const float*)d_in[8];   const float* fc21_b = (const float*)d_in[9];
  const float* fc22_W = (const float*)d_in[10];  const float* fc22_b = (const float*)d_in[11];
  const float* gc1_W  = (const float*)d_in[12];  const float* gc1_b  = (const float*)d_in[13];
  const float* gc2_W  = (const float*)d_in[14];  const float* gc2_b  = (const float*)d_in[15];
  const float* Wih0   = (const float*)d_in[16];  const float* Whh0   = (const float*)d_in[17];
  const float* bih0   = (const float*)d_in[18];  const float* bhh0   = (const float*)d_in[19];
  const float* Wih1   = (const float*)d_in[20];  const float* Whh1   = (const float*)d_in[21];
  const float* bih1   = (const float*)d_in[22];  const float* bhh1   = (const float*)d_in[23];
  const float* lfc_W  = (const float*)d_in[24];  const float* lfc_b  = (const float*)d_in[25];
  const float* fc_W   = (const float*)d_in[26];  const float* fc_b   = (const float*)d_in[27];

  float* out = (float*)d_out;                    // [64,12,512] then auto [64,24,512]

  // ---- workspace carve (floats) ----
  float* p = (float*)d_ws;
  float* a1      = p; p += 64 * 4096;            // relu(fc11)
  float* hidden  = p; p += 64 * 1024;            // relu(fc12)
  float* a3      = p; p += 64 * 4096;            // relu(fc21)
  float* auto_o  = p; p += 64 * 12288;           // relu(fc22) == autooutput
  float* feats   = p; p += 24 * 512 * 128;       // [S,N,2B]
  float* m1      = p; p += 24 * 512 * 128;
  float* T1      = p; p += 24 * 512 * 128;       // relu(gc1_W@m1) == h1^T per s
  float* m2      = p; p += 24 * 512 * 128;
  float* lstm_in = p; p += 24 * 128 * 256;       // out1 [S,2B,G2] == [64,24,512]
  float* gates   = p; p += 64 * 2048;
  float* hb0     = p; p += 64 * 512;
  float* cb0     = p; p += 64 * 512;
  float* hb1     = p; p += 64 * 512;
  float* cb1     = p; p += 64 * 512;
  float* hseq2   = p; p += 64 * 12 * 512;        // last PL steps of layer-1 h
  float* st_out  = p; p += 768 * 128;            // == [64,1536]
  float* nf      = p; p += 64 * 2560;

  // ---- 1) AutoEncoder (bandwidth-bound; streams ~440 MB of weights) ----
  gemm(stream, x, 12288, 0, fc11_W, 12288, 0, fc11_b, nullptr,
       a1, 4096, 0, 64, 4096, 12288, GEMM_OPB_T | GEMM_BIAS_N | GEMM_RELU, 1);
  gemm(stream, a1, 4096, 0, fc12_W, 4096, 0, fc12_b, nullptr,
       hidden, 1024, 0, 64, 1024, 4096, GEMM_OPB_T | GEMM_BIAS_N | GEMM_RELU, 1);
  gemm(stream, hidden, 1024, 0, fc21_W, 1024, 0, fc21_b, nullptr,
       a3, 4096, 0, 64, 4096, 1024, GEMM_OPB_T | GEMM_BIAS_N | GEMM_RELU, 1);
  gemm(stream, a3, 4096, 0, fc22_W, 4096, 0, fc22_b, nullptr,
       auto_o, 12288, 0, 64, 12288, 4096, GEMM_OPB_T | GEMM_BIAS_N | GEMM_RELU, 1);
  // autooutput is the second tuple output
  hipMemcpyAsync(out + 64 * 12 * 512, auto_o, (size_t)64 * 12288 * sizeof(float),
                 hipMemcpyDeviceToDevice, stream);

  // ---- 2) GCN block, batched over S=24 via gridDim.z ----
  k_build_feats<<<(24 * 512 * 128 + 255) / 256, 256, 0, stream>>>(x, auto_o, feats);
  // m1[s] = adj @ feats[s]
  gemm(stream, adj, 512, 0, feats, 128, 512 * 128, nullptr, nullptr,
       m1, 128, 512 * 128, 512, 128, 512, 0, 24);
  // T1[s] = relu(gc1_W @ m1[s] + gc1_b)   (bias per output row)
  gemm(stream, gc1_W, 512, 0, m1, 128, 512 * 128, gc1_b, nullptr,
       T1, 128, 512 * 128, 512, 128, 512, GEMM_RELU, 24);
  // m2[s] = adj @ T1[s]   (T1 == h1[s]^T, G1 == N == 512)
  gemm(stream, adj, 512, 0, T1, 128, 512 * 128, nullptr, nullptr,
       m2, 128, 512 * 128, 512, 128, 512, 0, 24);
  // out1[s,b,g] = (gc2_W @ m2[s])[g,b] + gc2_b[g]  -> transposed store
  gemm(stream, gc2_W, 512, 0, m2, 128, 512 * 128, gc2_b, nullptr,
       lstm_in, 256, 128 * 256, 256, 128, 512, GEMM_STORE_T, 24);

  // ---- 3) 2-layer LSTM (sequential over S; stream-ordered launches) ----
  hipMemcpyAsync(hb0, h0,             64 * 512 * sizeof(float), hipMemcpyDeviceToDevice, stream);
  hipMemcpyAsync(cb0, c0,             64 * 512 * sizeof(float), hipMemcpyDeviceToDevice, stream);
  hipMemcpyAsync(hb1, h0 + 64 * 512,  64 * 512 * sizeof(float), hipMemcpyDeviceToDevice, stream);
  hipMemcpyAsync(cb1, c0 + 64 * 512,  64 * 512 * sizeof(float), hipMemcpyDeviceToDevice, stream);

  for (int t = 0; t < 24; ++t) {
    const float* xt = lstm_in + t * 512;   // row b at b*12288 (== [64,24,512])
    // layer 0 gates = xt@Wih0^T + bih0 + h@Whh0^T + bhh0
    gemm(stream, xt, 12288, 0, Wih0, 512, 0, bih0, nullptr,
         gates, 2048, 0, 64, 2048, 512, GEMM_OPB_T | GEMM_BIAS_N, 1);
    gemm(stream, hb0, 512, 0, Whh0, 512, 0, bhh0, gates,
         gates, 2048, 0, 64, 2048, 512, GEMM_OPB_T | GEMM_BIAS_N | GEMM_ACCUM, 1);
    k_lstm_cell<<<128, 256, 0, stream>>>(gates, cb0, hb0, nullptr, 0);
    // layer 1, input is layer-0 h
    gemm(stream, hb0, 512, 0, Wih1, 512, 0, bih1, nullptr,
         gates, 2048, 0, 64, 2048, 512, GEMM_OPB_T | GEMM_BIAS_N, 1);
    gemm(stream, hb1, 512, 0, Whh1, 512, 0, bhh1, gates,
         gates, 2048, 0, 64, 2048, 512, GEMM_OPB_T | GEMM_BIAS_N | GEMM_ACCUM, 1);
    k_lstm_cell<<<128, 256, 0, stream>>>(gates, cb1, hb1,
                                         (t >= 12) ? hseq2 : nullptr, t - 12);
  }

  // ---- 4) lfc over last PL steps: [768,512] @ [512,128]^T -> st_out ----
  gemm(stream, hseq2, 512, 0, lfc_W, 512, 0, lfc_b, nullptr,
       st_out, 128, 0, 768, 128, 512, GEMM_OPB_T | GEMM_BIAS_N, 1);

  // ---- 5) final FC: concat(st_out, hidden) @ fc_W^T + fc_b -> out ----
  k_concat_nf<<<(64 * 2560 + 255) / 256, 256, 0, stream>>>(st_out, hidden, nf);
  gemm(stream, nf, 2560, 0, fc_W, 2560, 0, fc_b, nullptr,
       out, 6144, 0, 64, 6144, 2560, GEMM_OPB_T | GEMM_BIAS_N, 1);
}
